// RNADE_69234872812220
// MI455X (gfx1250) — matrix operand order repaired and checked by reading, not compiled, not run on hardware
//
#include <hip/hip_runtime.h>
#include <hip/hip_bf16.h>
#include <math.h>

#define B_SZ 16384
#define D_SZ 128
#define H_SZ 256
#define C_SZ 20

typedef __attribute__((ext_vector_type(16))) _Float16 v16h;
typedef __attribute__((ext_vector_type(8)))  float    v8f;
typedef unsigned short ushort_t;

// workspace layout (bytes)
#define VH_OFF 0ull                  // D * 4nt * 8kc * 512 halves  = 4 MB  (f16 B-fragments)
#define WF_OFF 4194304ull            // D * 256 floats              = 128 KB (W columns, A-layout order)
#define XT_OFF 4325376ull            // D * B floats                = 8 MB  (x transposed)

// A-operand (f16 16x32) K index for element e of v16h, given lane-half
__device__ __forceinline__ int ka_map(int half, int e) {
  return (e < 8) ? (half * 8 + e) : (16 + half * 8 + (e - 8));
}

// ---------------- pack kernels (one-time layout transforms) ----------------

// B-fragments: Vh[((d*4+nt)*8+kc)*512 + lane*16 + e] = Vcat[d][kc*32 + ka(lane>>4,e)][nt*16 + (lane&15)]
__global__ void pack_v(const float* __restrict__ Vp, const float* __restrict__ Vm,
                       const float* __restrict__ Vs, ushort_t* __restrict__ Vh) {
  int t = blockIdx.x * blockDim.x + threadIdx.x;   // D*4*8*32 = 131072 threads
  int lane = t & 31;
  int kc = (t >> 5) & 7;
  int nt = (t >> 8) & 3;
  int d = t >> 10;
  if (d >= D_SZ) return;
  int n  = nt * 16 + (lane & 15);
  int kb = kc * 32;
  _Float16* dst = (_Float16*)Vh + ((size_t)(d * 4 + nt) * 8 + kc) * 512 + lane * 16;
#pragma unroll
  for (int e = 0; e < 16; ++e) {
    int k = kb + ka_map(lane >> 4, e);
    float v = 0.f;
    if (n < 20)      v = Vp[((size_t)d * H_SZ + k) * C_SZ + n];
    else if (n < 40) v = Vm[((size_t)d * H_SZ + k) * C_SZ + (n - 20)];
    else if (n < 60) v = Vs[((size_t)d * H_SZ + k) * C_SZ + (n - 40)];
    dst[e] = (_Float16)v;
  }
}

// W column d in A-state order: Wf[d*256 + kc*32 + half*16 + e] = W[(kc*32 + ka(half,e)) * D + d]
__global__ void pack_w(const float* __restrict__ W, float* __restrict__ Wf) {
  int t = blockIdx.x * blockDim.x + threadIdx.x;   // D*8*2 = 2048 threads
  if (t >= D_SZ * 16) return;
  int half = t & 1;
  int kc = (t >> 1) & 7;
  int d = t >> 4;
  float* dst = Wf + (size_t)d * 256 + kc * 32 + half * 16;
#pragma unroll
  for (int e = 0; e < 16; ++e) {
    int k = kc * 32 + ka_map(half, e);
    dst[e] = W[(size_t)k * D_SZ + d];
  }
}

// xT[d*B + b] = x[b*D + d]   (B = 2^14)
__global__ void pack_x(const float* __restrict__ x, float* __restrict__ xT) {
  size_t t = (size_t)blockIdx.x * blockDim.x + threadIdx.x;
  if (t >= (size_t)D_SZ * B_SZ) return;
  int b = (int)(t & (B_SZ - 1));
  int d = (int)(t >> 14);
  xT[t] = x[(size_t)b * D_SZ + d];
}

// ---------------- main fused scan kernel ----------------

#if defined(__has_builtin)
#if __has_builtin(__builtin_amdgcn_global_load_async_to_lds_b128) && \
    __has_builtin(__builtin_amdgcn_s_wait_asynccnt)
#define USE_ASYNC_LDS 1
#endif
#endif

#ifdef USE_ASYNC_LDS
typedef int v4i_vs __attribute__((vector_size(16)));
typedef __attribute__((address_space(1))) v4i_vs gv4i;
typedef __attribute__((address_space(3))) v4i_vs lv4i;
#endif

__device__ __forceinline__ void stage(const ushort_t* __restrict__ Vh, int d,
                                      ushort_t* dstLds, int tid) {
  // 32 KB per step; 128 threads * 256 B each
  const char* src = (const char*)(Vh + (size_t)d * 16384) + tid * 256;
  char* dst = (char*)dstLds + tid * 256;
#ifdef USE_ASYNC_LDS
#pragma unroll
  for (int i = 0; i < 16; ++i) {
    __builtin_amdgcn_global_load_async_to_lds_b128(
        (gv4i*)(unsigned long long)(src + i * 16),
        (lv4i*)(unsigned)(unsigned long long)(dst + i * 16), 0, 0);
  }
#else
  const uint4* s4 = (const uint4*)src;
  uint4* d4 = (uint4*)dst;
#pragma unroll
  for (int i = 0; i < 16; ++i) d4[i] = s4[i];
#endif
}

__device__ __forceinline__ void wait_async() {
#ifdef USE_ASYNC_LDS
  __builtin_amdgcn_s_wait_asynccnt(0);
#endif
}

__device__ __forceinline__ v8f wmma_f16(v16h a, v16h b, v8f c) {
  return __builtin_amdgcn_wmma_f32_16x16x32_f16(false, a, false, b, (short)0, c,
                                                false, false);
}

__global__ __launch_bounds__(128) void rnade_main(
    const ushort_t* __restrict__ Vh, const float* __restrict__ Wf,
    const float* __restrict__ xT, const float* __restrict__ cvec,
    const float* __restrict__ bpi, const float* __restrict__ bmu,
    const float* __restrict__ bsg, float* __restrict__ out) {
  __shared__ __attribute__((aligned(64))) ushort_t Vbuf[2][16384];  // 64 KB ping-pong
  __shared__ float eps[4][64][20];                                  // per-wave epilogue scratch

  const int tid = threadIdx.x;
  const int lane = tid & 31;
  const int wave = tid >> 5;
  const int half = lane >> 4;
  const int nlo = lane & 15;
  const int rowbase = (blockIdx.x * 4 + wave) * 16;

  // a-state in registers, A-fragment order: areg[kc][e] = a[row=nlo][kc*32 + ka(half,e)]
  float areg[8][16];
#pragma unroll
  for (int kc = 0; kc < 8; ++kc)
#pragma unroll
    for (int e = 0; e < 16; ++e) areg[kc][e] = cvec[kc * 32 + ka_map(half, e)];

  float acc_logp = 0.f;

  stage(Vh, 0, &Vbuf[0][0], tid);

  for (int d = 0; d < D_SZ; ++d) {
    wait_async();
    __syncthreads();                                    // stage(d) visible; buf[d^1] free
    if (d + 1 < D_SZ) stage(Vh, d + 1, &Vbuf[(d + 1) & 1][0], tid);

    // ---- GEMM: s[16 rows][64 cols] = relu(a) @ Vcat[d] ----
    const v16h* vb16 = (const v16h*)&Vbuf[d & 1][0];
    v8f a0 = {}, a1 = {}, a2 = {}, a3 = {};
#pragma unroll
    for (int kc = 0; kc < 8; ++kc) {
      v16h ah;
#pragma unroll
      for (int e = 0; e < 16; ++e) ah[e] = (_Float16)fmaxf(areg[kc][e], 0.f);
      a0 = wmma_f16(ah, vb16[(0 * 8 + kc) * 32 + lane], a0);
      a1 = wmma_f16(ah, vb16[(1 * 8 + kc) * 32 + lane], a1);
      a2 = wmma_f16(ah, vb16[(2 * 8 + kc) * 32 + lane], a2);
      a3 = wmma_f16(ah, vb16[(3 * 8 + kc) * 32 + lane], a3);
    }

    // ---- spill C tiles to LDS (per-wave region; same-wave DS ops are in-order) ----
    {
      float(*ep)[20] = eps[wave];
      const int rb = half * 8;
#pragma unroll
      for (int e = 0; e < 8; ++e) {
        ep[0 * 16 + nlo][rb + e] = a0[e];
        ep[1 * 16 + nlo][rb + e] = a1[e];
        ep[2 * 16 + nlo][rb + e] = a2[e];
        ep[3 * 16 + nlo][rb + e] = a3[e];
      }
    }
    asm volatile("s_wait_dscnt 0" ::: "memory");

    // ---- mixture epilogue: 2 lanes per row, 10 components each ----
    {
      const int r = lane >> 1, pp = lane & 1;
      const float xv = xT[(size_t)d * B_SZ + rowbase + r];
      float num = 0.f, den = 0.f;
#pragma unroll
      for (int i = 0; i < 10; ++i) {
        int c = pp * 10 + i;
        float spi = eps[wave][c][r]      + bpi[d * C_SZ + c];
        float smu = eps[wave][20 + c][r] + bmu[d * C_SZ + c];
        float ssg = eps[wave][40 + c][r] + bsg[d * C_SZ + c];
        float epi = __expf(spi);
        float sig = __expf(ssg);
        float is  = __builtin_amdgcn_rcpf(sig);
        float z   = (xv - smu) * is;
        float g   = 0.3989422804014327f * is * __expf(-0.5f * z * z);
        num += epi * g;
        den += epi;
      }
      num += __shfl_xor(num, 1, 32);
      den += __shfl_xor(den, 1, 32);
      acc_logp += __logf(num * __builtin_amdgcn_rcpf(den) + 1e-10f);
    }

    // ---- rank-1 state update: a += x_d * W[:,d] ----
    {
      const float xm = xT[(size_t)d * B_SZ + rowbase + nlo];
      const float* wf = Wf + (size_t)d * 256 + half * 16;
#pragma unroll
      for (int kc = 0; kc < 8; ++kc)
#pragma unroll
        for (int e = 0; e < 16; ++e)
          areg[kc][e] = fmaf(xm, wf[kc * 32 + e], areg[kc][e]);
    }
  }

  if ((lane & 1) == 0) out[rowbase + (lane >> 1)] = acc_logp;
}

// ---------------- launch ----------------

extern "C" void kernel_launch(void* const* d_in, const int* in_sizes, int n_in,
                              void* d_out, int out_size, void* d_ws, size_t ws_size,
                              hipStream_t stream) {
  const float* x   = (const float*)d_in[0];
  const float* W   = (const float*)d_in[1];
  const float* c   = (const float*)d_in[2];
  const float* bpi = (const float*)d_in[3];
  const float* Vpi = (const float*)d_in[4];
  const float* bmu = (const float*)d_in[5];
  const float* Vmu = (const float*)d_in[6];
  const float* bsg = (const float*)d_in[7];
  const float* Vsg = (const float*)d_in[8];
  float* out = (float*)d_out;
  char* ws = (char*)d_ws;
  ushort_t* Vh = (ushort_t*)(ws + VH_OFF);
  float* Wf = (float*)(ws + WF_OFF);
  float* xT = (float*)(ws + XT_OFF);

  pack_v<<<512, 256, 0, stream>>>(Vpi, Vmu, Vsg, Vh);
  pack_w<<<8, 256, 0, stream>>>(W, Wf);
  pack_x<<<(D_SZ * B_SZ) / 256, 256, 0, stream>>>(x, xT);
  rnade_main<<<256, 128, 0, stream>>>(Vh, Wf, xT, c, bpi, bmu, bsg, out);
}
